// TOF_CRNNcell_8916352106614
// MI455X (gfx1250) — matrix operand order repaired and checked by reading, not compiled, not run on hardware
//
#include <hip/hip_runtime.h>
#include <math.h>

// MI455X / gfx1250 implementation.
// All convs are implicit GEMMs on v_wmma_f32_16x16x32_f16 (f16 in, f32 acc).
// Weights pre-packed to the WMMA A-matrix lane layout; activations staged to
// LDS channel-contiguous so B fragments are single 32B ds loads per lane.
// DCN bilinear gather is fused into its GEMM via an NHWC f16 copy of x.

typedef __attribute__((ext_vector_type(16))) _Float16 v16h;
typedef __attribute__((ext_vector_type(8)))  float    v8f;

#define HH 128
#define WW 128
#define NFEAT (4 * 64 * 128 * 128)

// ---------------------------------------------------------------------------
// Pack fp32 OIHW 3x3 weights into WMMA 16-bit A-matrix fragments.
// K ordering: K = tap*Cin + cin. Output: [CoutTile16][Kchunk32][lane32][half16].
// A order used here (16-bit A 16x32; not the ISA table order): M = lane%16,
//   kk = (lane<16?0:16) + half   (K-contiguous per lane, matching B)
// ---------------------------------------------------------------------------
__global__ __launch_bounds__(256) void pack_w_kernel(const float* __restrict__ w,
                                                     _Float16* __restrict__ wp,
                                                     int Cout, int Cin) {
  int nChunks = (Cin * 9) / 32;
  int total = ((Cout + 15) / 16) * nChunks * 512;
  int i = blockIdx.x * 256 + threadIdx.x;
  if (i >= total) return;
  int hIdx = i & 15;
  int lane = (i >> 4) & 31;
  int rest = i >> 9;
  int chunk = rest % nChunks;
  int ct = rest / nChunks;
  int v = hIdx >> 1, s = hIdx & 1;
  int kk = ((lane < 16) ? 0 : 16) + hIdx;
  int K = chunk * 32 + kk;
  int tap = K / Cin, cin = K % Cin;
  int cout = ct * 16 + (lane & 15);
  _Float16 val = (_Float16)0.f;
  if (cout < Cout) val = (_Float16)w[(cout * Cin + cin) * 9 + tap];
  wp[i] = val;
}

// ---------------------------------------------------------------------------
// NCHW fp32 -> NHWC f16 cast of x (for channel-coalesced DCN gathers).
// ---------------------------------------------------------------------------
__global__ __launch_bounds__(256) void cast_nhwc_kernel(const float* __restrict__ src,
                                                        _Float16* __restrict__ dst) {
  int i = blockIdx.x * 256 + threadIdx.x;  // NHWC linear
  if (i >= NFEAT) return;
  int c = i & 63;
  int rest = i >> 6;
  int w = rest & 127;
  rest >>= 7;
  int h = rest & 127;
  int b = rest >> 7;
  dst[i] = (_Float16)src[((b * 64 + c) * HH + h) * WW + w];
}

// ---------------------------------------------------------------------------
// Generic 3x3 SAME conv as implicit GEMM via WMMA.
// Workgroup: one row, 32 pixels, all Cout. 8 waves = (pixelTile 0/1) x CoutTiles.
// src = concat(src0[0:split], src1[split:Cin]) along channels (pass split=Cin
// and src1=src0 for non-concat convs). act: 0 none, 1 lrelu(0.1), 2 relu.
// ---------------------------------------------------------------------------
__global__ __launch_bounds__(256) void conv3x3_wmma_kernel(
    const float* __restrict__ src0, const float* __restrict__ src1, int split,
    int Cin, int Cout, const _Float16* __restrict__ wp,
    const float* __restrict__ bias, float* __restrict__ dst, int act) {
  __shared__ _Float16 smem[3 * 34 * 128];  // [row][col][cin] halves, 26 KB max
  int w0 = blockIdx.x * 32;
  int h = blockIdx.y;
  int b = blockIdx.z;
  int tid = threadIdx.x;

  // Stage halo: rows h-1..h+1, cols w0-1..w0+32, channel-contiguous f16.
  for (int i = tid; i < Cin * 102; i += 256) {
    int cin = i / 102;
    int rem = i - cin * 102;
    int r = rem / 34;
    int col = rem - r * 34;
    int hh = h - 1 + r;
    int ww = w0 - 1 + col;
    float v = 0.f;
    if (hh >= 0 && hh < HH && ww >= 0 && ww < WW) {
      v = (cin < split)
            ? src0[((b * split + cin) * HH + hh) * WW + ww]
            : src1[((b * (Cin - split) + (cin - split)) * HH + hh) * WW + ww];
    }
    smem[(r * 34 + col) * Cin + cin] = (_Float16)v;
  }
  __syncthreads();

  int wave = tid >> 5;
  int lane = tid & 31;
  int nCT = (Cout + 15) >> 4;
  if (wave < nCT * 2) {
    int pt = wave / nCT;     // pixel tile (16 px each)
    int ct = wave % nCT;     // cout tile
    int n0 = pt * 16;
    int ccN = Cin >> 5;      // 32-wide K chunks per tap
    const v16h* wpv = (const v16h*)wp;
    v8f acc = {};
    for (int tap = 0; tap < 9; ++tap) {
      int dy = tap / 3, dx = tap % 3;
      for (int cc = 0; cc < ccN; ++cc) {
        int chunk = tap * ccN + cc;
        v16h a = wpv[(ct * (9 * ccN) + chunk) * 32 + lane];
        // B 32x16 order used here: N=lane%16; lanes<16 hold K 0..15, lanes>=16
        // hold K 16..31, K-contiguous per lane -> one aligned 32B LDS read.
        int col = n0 + (lane & 15) + dx;  // staged col (pad already applied)
        const v16h* bp =
            (const v16h*)&smem[(dy * 34 + col) * Cin + cc * 32 + ((lane < 16) ? 0 : 16)];
        acc = __builtin_amdgcn_wmma_f32_16x16x32_f16(false, a, false, *bp,
                                                     (short)0, acc, false, false);
      }
    }
    // C/D layout: VGPR r -> M = r (+8 for lanes 16..31), N = lane%16.
    int n = lane & 15;
    int mofs = (lane < 16) ? 0 : 8;
    for (int r = 0; r < 8; ++r) {
      int cout = ct * 16 + r + mofs;
      if (cout < Cout) {
        float v = acc[r] + bias[cout];
        if (act == 1) v = (v >= 0.f) ? v : 0.1f * v;
        else if (act == 2) v = fmaxf(v, 0.f);
        dst[((b * Cout + cout) * HH + h) * WW + w0 + n0 + n] = v;
      }
    }
  }
}

// ---------------------------------------------------------------------------
// Deformable-conv core fused with its GEMM (M=64, N=32 px, K=C*9=576).
// Per-pixel/tap bilinear metadata precomputed to LDS; each K-chunk (tap, 32ch)
// is gathered from NHWC f16 x and written directly in B-fragment layout.
// ---------------------------------------------------------------------------
__global__ __launch_bounds__(256) void dcn_wmma_kernel(
    const _Float16* __restrict__ xh, const float* __restrict__ om,
    const _Float16* __restrict__ wp, const float* __restrict__ bias,
    float* __restrict__ dst) {
  __shared__ int sY0[288], sY1[288], sX0[288], sX1[288];
  __shared__ float sW00[288], sW01[288], sW10[288], sW11[288];
  __shared__ _Float16 sB[2 * 32 * 16];  // [pixelTile][lane][half]
  const int C = 64;
  int w0 = blockIdx.x * 32;
  int h = blockIdx.y;
  int b = blockIdx.z;
  int tid = threadIdx.x;

  // Precompute bilinear corners + (mask * validity)-folded weights.
  for (int i = tid; i < 288; i += 256) {
    int n = i & 31;
    int t = i >> 5;
    int ky = t / 3, kx = t % 3;
    int w = w0 + n;
    int base = ((b * 27) * HH + h) * WW + w;
    float oy = om[base + t * HH * WW];
    float ox = om[base + (9 + t) * HH * WW];
    float mk = om[base + (18 + t) * HH * WW];
    float mask = 1.f / (1.f + __expf(-mk));
    float py = (float)(h - 1 + ky) + oy;
    float px = (float)(w - 1 + kx) + ox;
    float y0f = floorf(py), x0f = floorf(px);
    float ly = py - y0f, lx = px - x0f;
    float y1f = y0f + 1.f, x1f = x0f + 1.f;
    bool vy0 = (y0f >= 0.f) && (y0f <= 127.f);
    bool vy1 = (y1f >= 0.f) && (y1f <= 127.f);
    bool vx0 = (x0f >= 0.f) && (x0f <= 127.f);
    bool vx1 = (x1f >= 0.f) && (x1f <= 127.f);
    sW00[i] = (1.f - ly) * (1.f - lx) * mask * ((vy0 && vx0) ? 1.f : 0.f);
    sW01[i] = (1.f - ly) * lx * mask * ((vy0 && vx1) ? 1.f : 0.f);
    sW10[i] = ly * (1.f - lx) * mask * ((vy1 && vx0) ? 1.f : 0.f);
    sW11[i] = ly * lx * mask * ((vy1 && vx1) ? 1.f : 0.f);
    sY0[i] = min(max((int)y0f, 0), 127);
    sY1[i] = min(max((int)y1f, 0), 127);
    sX0[i] = min(max((int)x0f, 0), 127);
    sX1[i] = min(max((int)x1f, 0), 127);
  }
  __syncthreads();

  int wave = tid >> 5;
  int lane = tid & 31;
  int pt = wave >> 2;  // pixel tile
  int ct = wave & 3;   // cout tile
  const v16h* wpv = (const v16h*)wp;
  v8f acc = {};
  int bn = tid & 31;  // builder pixel
  int cq = tid >> 5;  // builder channel quad (x4 -> 32 channels)

  for (int ch = 0; ch < 18; ++ch) {  // K chunks: (tap, half-of-64-channels)
    int t = ch >> 1;
    int c0 = (ch & 1) * 32;
    int mi = t * 32 + bn;
    int y0 = sY0[mi], y1 = sY1[mi], x0 = sX0[mi], x1 = sX1[mi];
    float w00 = sW00[mi], w01 = sW01[mi], w10 = sW10[mi], w11 = sW11[mi];
    int p00 = ((b * HH + y0) * WW + x0) * C;
    int p01 = ((b * HH + y0) * WW + x1) * C;
    int p10 = ((b * HH + y1) * WW + x0) * C;
    int p11 = ((b * HH + y1) * WW + x1) * C;
    int ptb = bn >> 4;
    int N = bn & 15;
    for (int j = 0; j < 4; ++j) {
      int kk = cq * 4 + j;
      int c = c0 + kk;
      float v = w00 * (float)xh[p00 + c] + w01 * (float)xh[p01 + c] +
                w10 * (float)xh[p10 + c] + w11 * (float)xh[p11 + c];
      int ln = N + ((kk < 16) ? 0 : 16);
      sB[(ptb * 32 + ln) * 16 + (kk & 15)] = (_Float16)v;
    }
    __syncthreads();
    {
      v16h a = wpv[(ct * 18 + ch) * 32 + lane];
      const v16h* bp = (const v16h*)&sB[(pt * 32 + lane) * 16];
      acc = __builtin_amdgcn_wmma_f32_16x16x32_f16(false, a, false, *bp,
                                                   (short)0, acc, false, false);
    }
    __syncthreads();
  }

  int n = lane & 15;
  int mofs = (lane < 16) ? 0 : 8;
  for (int r = 0; r < 8; ++r) {
    int cout = ct * 16 + r + mofs;
    float v = acc[r] + bias[cout];
    dst[((b * 64 + cout) * HH + h) * WW + w0 + pt * 16 + n] = v;
  }
}

// ---------------------------------------------------------------------------
extern "C" void kernel_launch(void* const* d_in, const int* in_sizes, int n_in,
                              void* d_out, int out_size, void* d_ws, size_t ws_size,
                              hipStream_t stream) {
  const float* x        = (const float*)d_in[0];
  const float* x_neigh  = (const float*)d_in[1];
  const float* offset_t = (const float*)d_in[2];
  const float* w_off1 = (const float*)d_in[3];
  const float* b_off1 = (const float*)d_in[4];
  const float* w_off2 = (const float*)d_in[5];
  const float* b_off2 = (const float*)d_in[6];
  const float* w_off3 = (const float*)d_in[7];
  const float* b_off3 = (const float*)d_in[8];
  const float* w_om   = (const float*)d_in[9];
  const float* b_om   = (const float*)d_in[10];
  const float* w_dcn  = (const float*)d_in[11];
  const float* b_dcn  = (const float*)d_in[12];
  const float* w_out  = (const float*)d_in[13];
  const float* b_out  = (const float*)d_in[14];

  char* ws = (char*)d_ws;
  float*    t0  = (float*)(ws + (size_t)(0u));          // 16 MB
  float*    t1  = (float*)(ws + ((size_t)16u << 20));   // 16 MB
  float*    om  = (float*)(ws + ((size_t)32u << 20));   // 7 MB
  _Float16* xh  = (_Float16*)(ws + ((size_t)40u << 20));// 8 MB
  float*    al  = (float*)(ws + ((size_t)48u << 20));   // 16 MB
  _Float16* wp1 = (_Float16*)(ws + ((size_t)64u << 20) + 0u * 262144u);
  _Float16* wp2 = (_Float16*)(ws + ((size_t)64u << 20) + 1u * 262144u);
  _Float16* wp3 = (_Float16*)(ws + ((size_t)64u << 20) + 2u * 262144u);
  _Float16* wpm = (_Float16*)(ws + ((size_t)64u << 20) + 3u * 262144u);
  _Float16* wpd = (_Float16*)(ws + ((size_t)64u << 20) + 4u * 262144u);
  _Float16* wpo = (_Float16*)(ws + ((size_t)64u << 20) + 5u * 262144u);

  float* align_out  = (float*)d_out;           // output 0
  float* offset_out = (float*)d_out + NFEAT;   // output 1 (lrelu(conv3))

  // Weight packing (tiny kernels).
  auto packBlocks = [](int Cout, int Cin) {
    int total = ((Cout + 15) / 16) * ((Cin * 9) / 32) * 512;
    return (total + 255) / 256;
  };
  pack_w_kernel<<<packBlocks(64, 128), 256, 0, stream>>>(w_off1, wp1, 64, 128);
  pack_w_kernel<<<packBlocks(64, 128), 256, 0, stream>>>(w_off2, wp2, 64, 128);
  pack_w_kernel<<<packBlocks(64, 64),  256, 0, stream>>>(w_off3, wp3, 64, 64);
  pack_w_kernel<<<packBlocks(27, 64),  256, 0, stream>>>(w_om,   wpm, 27, 64);
  pack_w_kernel<<<packBlocks(64, 64),  256, 0, stream>>>(w_dcn,  wpd, 64, 64);
  pack_w_kernel<<<packBlocks(64, 64),  256, 0, stream>>>(w_out,  wpo, 64, 64);

  // NHWC f16 copy of x for the DCN gathers.
  cast_nhwc_kernel<<<NFEAT / 256, 256, 0, stream>>>(x, xh);

  dim3 grid(WW / 32, HH, 4);
  // offset_1 = lrelu(conv([x, x_neigh]))
  conv3x3_wmma_kernel<<<grid, 256, 0, stream>>>(x, x_neigh, 64, 128, 64, wp1,
                                                b_off1, t0, 1);
  // offset_2 = lrelu(conv([offset_1, offset_t]))
  conv3x3_wmma_kernel<<<grid, 256, 0, stream>>>(t0, offset_t, 64, 128, 64, wp2,
                                                b_off2, t1, 1);
  // offset = lrelu(conv(offset_2))  -> second output slot
  conv3x3_wmma_kernel<<<grid, 256, 0, stream>>>(t1, t1, 64, 64, 64, wp3,
                                                b_off3, offset_out, 1);
  // om = conv(offset), 27 channels, no activation
  conv3x3_wmma_kernel<<<grid, 256, 0, stream>>>(offset_out, offset_out, 64, 64,
                                                27, wpm, b_om, om, 0);
  // align_feats = dcn_core(x, offsets, sigmoid(mask))
  dcn_wmma_kernel<<<grid, 256, 0, stream>>>(xh, om, wpd, b_dcn, al);
  // out = relu(conv(align_feats)) -> first output slot
  conv3x3_wmma_kernel<<<grid, 256, 0, stream>>>(al, al, 64, 64, 64, wpo,
                                                b_out, align_out, 2);
}
